// MSTR_I_65317862637875
// MI455X (gfx1250) — matrix operand order repaired and checked by reading, not compiled, not run on hardware
//
#include <hip/hip_runtime.h>
#include <hip/hip_bf16.h>

// ---------------------------------------------------------------------------
// Sizes from the reference
// ---------------------------------------------------------------------------
#define NB   2048   // batch N
#define TT_  256    // time T
#define FF   16     // input features F
#define DD   128    // model dim D
#define SS   2048   // tickers S

typedef __attribute__((ext_vector_type(16))) _Float16 v16h;
typedef __attribute__((ext_vector_type(8)))  _Float16 v8h;
typedef __attribute__((ext_vector_type(8)))  float    v8f;

#define WMMA_F16(A, B, C) \
  __builtin_amdgcn_wmma_f32_16x16x32_f16(false, (A), false, (B), (short)0, (C), false, false)

// ---------------------------------------------------------------------------
// Fragment loaders (CDNA5 wave32 WMMA layouts, CDNA5 ISA §7.12.2)
//
// A matrix 16x32 f16 (M x K), source row-major with leading dim `ld`:
//   lane<16 : row=lane,    K chunks [0..7] and [16..23]
//   lane>=16: row=lane-16, K chunks [8..15] and [24..31]
// -> two contiguous 16B loads per lane.
__device__ __forceinline__ v16h ld_fragA(const _Float16* __restrict__ p, int ld) {
  const int lane = threadIdx.x & 31;
  const _Float16* r = p + (lane & 15) * ld + ((lane >> 4) << 3);
  union { v16h v; v8h h[2]; } f;
  f.h[0] = *(const v8h*)(r);
  f.h[1] = *(const v8h*)(r + 16);
  return f.v;
}

// B matrix 32x16 f16 (K x N), source column-major ("[n][k]" with leading dim ld):
//   lane<16 : col=lane,    K rows [0..7] and [16..23]
//   lane>=16: col=lane-16, K rows [8..15] and [24..31]
// -> two contiguous 16B loads per lane.
__device__ __forceinline__ v16h ld_fragB(const _Float16* p, int ld) {
  const int lane = threadIdx.x & 31;
  const _Float16* r = p + (lane & 15) * ld + ((lane >> 4) << 3);
  union { v16h v; v8h h[2]; } f;
  f.h[0] = *(const v8h*)(r);
  f.h[1] = *(const v8h*)(r + 16);
  return f.v;
}

// ---------------------------------------------------------------------------
// Weight conversion fp32 -> f16 with layout transform / zero padding
// ---------------------------------------------------------------------------
// src (O, C, 3) row-major  ->  dst [k][o][CP]  (c >= C zero padded)
__global__ void conv_w_to_f16(const float* __restrict__ src, _Float16* __restrict__ dst,
                              int O, int C, int CP) {
  int i = blockIdx.x * blockDim.x + threadIdx.x;
  int total = 3 * O * CP;
  if (i >= total) return;
  int k = i / (O * CP);
  int r = i - k * (O * CP);
  int o = r / CP;
  int c = r - o * CP;
  float v = (c < C) ? src[(o * C + c) * 3 + k] : 0.0f;
  dst[i] = (_Float16)v;
}

// src (O, C) row-major -> dst (O, CP), zero pad c >= C
__global__ void mat_to_f16(const float* __restrict__ src, _Float16* __restrict__ dst,
                           int O, int C, int CP) {
  int i = blockIdx.x * blockDim.x + threadIdx.x;
  if (i >= O * CP) return;
  int o = i / CP;
  int c = i - o * CP;
  dst[i] = (_Float16)((c < C) ? src[o * C + c] : 0.0f);
}

// ---------------------------------------------------------------------------
// Self-attention: per-n (T=256, F=16), online softmax. Output f16 (n, t, 16).
// ---------------------------------------------------------------------------
__global__ __launch_bounds__(256)
void attn_kernel(const float* __restrict__ x, _Float16* __restrict__ att) {
  const int n = blockIdx.x;
  const int t = threadIdx.x;
  __shared__ float xs[TT_][FF];
  const float* xr = x + ((size_t)n * TT_ + t) * FF;
#pragma unroll
  for (int f = 0; f < FF; ++f) xs[t][f] = xr[f];
  __syncthreads();

  float myx[FF];
#pragma unroll
  for (int f = 0; f < FF; ++f) myx[f] = xs[t][f];

  float m = -1e30f, sum = 0.0f;
  float acc[FF];
#pragma unroll
  for (int f = 0; f < FF; ++f) acc[f] = 0.0f;

  for (int s = 0; s < TT_; ++s) {
    float d = 0.0f;
#pragma unroll
    for (int f = 0; f < FF; ++f) d += myx[f] * xs[s][f];
    d *= 0.25f;  // 1/sqrt(16)
    float mn = fmaxf(m, d);
    float sc = __expf(m - mn);
    float p  = __expf(d - mn);
    sum = sum * sc + p;
#pragma unroll
    for (int f = 0; f < FF; ++f) acc[f] = acc[f] * sc + p * xs[s][f];
    m = mn;
  }
  float inv = 1.0f / sum;
  union { v8h h; _Float16 e[8]; } o0, o1;
#pragma unroll
  for (int f = 0; f < 8; ++f) o0.e[f] = (_Float16)(myx[f] + acc[f] * inv);
#pragma unroll
  for (int f = 0; f < 8; ++f) o1.e[f] = (_Float16)(myx[8 + f] + acc[8 + f] * inv);
  _Float16* op = att + ((size_t)n * TT_ + t) * FF;
  *(v8h*)(op)     = o0.h;
  *(v8h*)(op + 8) = o1.h;
}

// ---------------------------------------------------------------------------
// Fused TCN layer: causal dilated conv (K=3) + GLU + residual, all via WMMA.
// Activations (n, t, c) f16. Block = (n, 64-wide t tile), 8 waves,
// wave w owns output-channel tile w (16 channels).
// Input tile staged into LDS with GLOBAL_LOAD_ASYNC_TO_LDS (ASYNCcnt path)
// for the 128-channel layers.
// ---------------------------------------------------------------------------
template <int CIN, int CPAD, int DIL, bool DSRES>
__global__ __launch_bounds__(256)
void tcn_layer_kernel(const _Float16* __restrict__ in, _Float16* __restrict__ out,
                      const _Float16* __restrict__ convW, const float* __restrict__ convB,
                      const _Float16* __restrict__ g1W,   const float* __restrict__ g1B,
                      const _Float16* __restrict__ g2W,   const float* __restrict__ g2B,
                      const _Float16* __restrict__ dsW,   const float* __restrict__ dsB) {
  constexpr int TTILE  = 64;
  constexpr int HALO   = 2 * DIL;
  constexpr int INROWS = TTILE + HALO;
  constexpr int LDIN   = CPAD + 8;   // f16 pitch -> 4*row mod 64 bank stride
  constexpr int LDH    = 128 + 8;
  constexpr int KCH    = CPAD / 32;  // K chunks of 32 for the conv

  __shared__ __align__(16) _Float16 sin_t[INROWS * LDIN]; // input tile, [t][c]
  __shared__ __align__(16) _Float16 sh_t[TTILE * LDH];    // conv output h, [t][c]

  const int nb   = blockIdx.x >> 2;
  const int t0   = (blockIdx.x & 3) * TTILE;
  const int tid  = threadIdx.x;
  const int lane = tid & 31;
  const int wave = tid >> 5;
  const int ot   = wave;             // o-tile 0..7

  // L2 should keep the (tiny) weights hot; hint the prefetcher anyway.
  __builtin_prefetch(g1W + tid * 8, 0, 0);
  __builtin_prefetch(g2W + tid * 8, 0, 0);

  // ---- stage input tile (with causal halo, zero pad t<0 and c>=CIN) ----
  const size_t inbase = (size_t)nb * TT_ * CIN;
  if constexpr (CIN == CPAD) {
    // zero-fill causal-pad rows (only blocks with t0==0 have any)
    {
      v8h z = {};
      for (int e = tid * 8; e < HALO * CPAD; e += 256 * 8) {
        int r  = e / CPAD;
        int cc = e - r * CPAD;
        if (t0 - HALO + r < 0) *(v8h*)&sin_t[r * LDIN + cc] = z;
      }
    }
    // async copy of valid rows: one issue moves 2 rows (32 lanes x 16B),
    // tracked by ASYNCcnt, no VGPR data path.
    const int col = lane & 15;
    for (int p = wave * 2; p < INROWS; p += 16) {
      const int r = p + (lane >> 4);
      const int t = t0 - HALO + r;
      if (t >= 0) {
        const _Float16* gsrc = in + inbase + (size_t)t * CIN + col * 8;
        unsigned ldsa = (unsigned)(size_t)&sin_t[r * LDIN + col * 8];
        asm volatile("global_load_async_to_lds_b128 %0, %1, off"
                     :: "v"(ldsa), "v"(gsrc) : "memory");
      }
    }
    asm volatile("s_wait_asynccnt 0x0" ::: "memory");
  } else {
    for (int e = tid * 8; e < INROWS * CPAD; e += 256 * 8) {
      int r  = e / CPAD;
      int cc = e - r * CPAD;
      int t  = t0 - HALO + r;
      v8h val = {};
      if (t >= 0 && cc < CIN) val = *(const v8h*)(in + inbase + (size_t)t * CIN + cc);
      *(v8h*)&sin_t[r * LDIN + cc] = val;
    }
  }
  __syncthreads();

  const int g     = lane >> 4;
  const int obase = ot * 16 + g * 8;

  // ---- conv phase: h[o][t] = sum_k sum_c Wk[o][c] * in[t-(2-k)*DIL][c] ----
  // All 4 t-tile accumulators live simultaneously; per (k,kk) chunk load A
  // once and 4 B fragments into distinct registers so ds loads clause up and
  // overlap the XDL pipe instead of ds_load->wait->wmma lockstep.
  {
    v8f acc[4] = {};
#pragma unroll
    for (int k = 0; k < 3; ++k) {
      const int shift = HALO - (2 - k) * DIL;  // 0, DIL, 2*DIL
      const _Float16* abase = convW + (size_t)(k * 128 + ot * 16) * CPAD;
#pragma unroll
      for (int kk = 0; kk < KCH; ++kk) {
        v16h A  = ld_fragA(abase + kk * 32, CPAD);
        v16h B0 = ld_fragB(&sin_t[(0 * 16 + shift) * LDIN + kk * 32], LDIN);
        v16h B1 = ld_fragB(&sin_t[(1 * 16 + shift) * LDIN + kk * 32], LDIN);
        v16h B2 = ld_fragB(&sin_t[(2 * 16 + shift) * LDIN + kk * 32], LDIN);
        v16h B3 = ld_fragB(&sin_t[(3 * 16 + shift) * LDIN + kk * 32], LDIN);
        acc[0] = WMMA_F16(A, B0, acc[0]);
        acc[1] = WMMA_F16(A, B1, acc[1]);
        acc[2] = WMMA_F16(A, B2, acc[2]);
        acc[3] = WMMA_F16(A, B3, acc[3]);
      }
    }
#pragma unroll
    for (int tt = 0; tt < 4; ++tt) {
      v8h hv;
#pragma unroll
      for (int r = 0; r < 8; ++r) hv[r] = (_Float16)(acc[tt][r] + convB[obase + r]);
      *(v8h*)&sh_t[(tt * 16 + (lane & 15)) * LDH + obase] = hv;
    }
  }
  __syncthreads();

  // ---- GLU + residual phase ----
  const size_t outbase = (size_t)nb * TT_ * 128;
  v8f aacc[4] = {}, gacc[4] = {};
#pragma unroll
  for (int kk = 0; kk < 4; ++kk) {
    v16h A1 = ld_fragA(g1W + (size_t)(ot * 16) * 128 + kk * 32, 128);
    v16h A2 = ld_fragA(g2W + (size_t)(ot * 16) * 128 + kk * 32, 128);
#pragma unroll
    for (int tt = 0; tt < 4; ++tt) {
      v16h B = ld_fragB(&sh_t[(tt * 16) * LDH + kk * 32], LDH);
      aacc[tt] = WMMA_F16(A1, B, aacc[tt]);
      gacc[tt] = WMMA_F16(A2, B, gacc[tt]);
    }
  }
#pragma unroll
  for (int tt = 0; tt < 4; ++tt) {
    const int trow = tt * 16 + (lane & 15);
    v8f racc = {};
    v8h res16 = {};
    if constexpr (DSRES) {
      // 16->128 downsample residual: one padded-K WMMA against the input tile
      v16h Ad = ld_fragA(dsW + (size_t)(ot * 16) * CPAD, CPAD);
      v16h Bi = ld_fragB(&sin_t[(HALO + tt * 16) * LDIN], LDIN);
      racc = WMMA_F16(Ad, Bi, racc);
    } else {
      res16 = *(const v8h*)&sin_t[(HALO + trow) * LDIN + obase];
    }
    v8h ov;
#pragma unroll
    for (int r = 0; r < 8; ++r) {
      float a  = aacc[tt][r] + g1B[obase + r];
      float gg = gacc[tt][r] + g2B[obase + r];
      float h  = a * (1.0f / (1.0f + __expf(-gg)));
      float rr;
      if constexpr (DSRES) rr = racc[r] + dsB[obase + r];
      else                 rr = (float)res16[r];
      ov[r] = (_Float16)(h + rr);
    }
    *(v8h*)&out[outbase + (size_t)(t0 + trow) * 128 + obase] = ov;
  }
}

// ---------------------------------------------------------------------------
// Hawkes head (per n): q = lin_in( emb[idx] @ wq^T + bq ), softmax attention
// over context, exp-decay mix, concat, tanh(lin_out).
// ---------------------------------------------------------------------------
__global__ __launch_bounds__(256)
void hawkes_kernel(const _Float16* __restrict__ ctx, const int* __restrict__ idx,
                   const float* __restrict__ emb,  const float* __restrict__ wq,
                   const float* __restrict__ bq,   const float* __restrict__ linin,
                   const float* __restrict__ linout,
                   const float* __restrict__ aeArr, const float* __restrict__ abArr,
                   float* __restrict__ out) {
  const int n = blockIdx.x;
  const int tid = threadIdx.x;
  __shared__ float e[DD], q0[DD], q[DD], sc[TT_], aw[TT_], comb[2 * DD];

  const int tk = idx[n];
  if (tid < DD) e[tid] = emb[(size_t)tk * DD + tid];
  __syncthreads();

  if (tid < DD) {
    float a = bq[tid];
    for (int d = 0; d < DD; ++d) a += e[d] * wq[tid * DD + d];
    q0[tid] = a;
  }
  __syncthreads();
  if (tid < DD) {
    float a = 0.0f;
    for (int d = 0; d < DD; ++d) a += q0[d] * linin[tid * DD + d];
    q[tid] = a;
  }
  __syncthreads();

  const _Float16* c = ctx + (size_t)n * TT_ * DD;
  {
    float s = 0.0f;
    for (int d = 0; d < DD; ++d) s += q[d] * (float)c[tid * DD + d];
    sc[tid] = s;
  }
  __syncthreads();
  float m = -1e30f;
  for (int s = 0; s < TT_; ++s) m = fmaxf(m, sc[s]);
  aw[tid] = __expf(sc[tid] - m);
  __syncthreads();
  float sum = 0.0f;
  for (int s = 0; s < TT_; ++s) sum += aw[s];
  const float inv = 1.0f / sum;

  const float ae = aeArr[tk];
  const float ab = abArr[tk];
  if (tid < DD) {
    const int d = tid;
    float acc = 0.0f;
    for (int t = 0; t < TT_; ++t) {
      float cv  = (float)c[t * DD + d];
      float mix = aw[t] * inv * cv;
      float bt  = __expf(-ab * (float)(TT_ - 1 - t));
      float t2  = fmaxf(ae * mix * bt, 0.0f);
      acc += t2 + mix;
    }
    comb[d] = acc;
  } else {
    comb[DD + (tid - DD)] = q[tid - DD];
  }
  __syncthreads();
  if (tid < DD) {
    float a = 0.0f;
    for (int cc = 0; cc < 2 * DD; ++cc) a += comb[cc] * linout[tid * 2 * DD + cc];
    out[(size_t)n * DD + tid] = tanhf(a);
  }
}

// ---------------------------------------------------------------------------
// Host launcher
// ---------------------------------------------------------------------------
extern "C" void kernel_launch(void* const* d_in, const int* in_sizes, int n_in,
                              void* d_out, int out_size, void* d_ws, size_t ws_size,
                              hipStream_t stream) {
  (void)in_sizes; (void)n_in; (void)out_size; (void)ws_size;

  const float* x   = (const float*)d_in[0];
  const int*   idx = (const int*)d_in[1];
  // params flattened in setup_inputs insertion order
  const float* conv_w[4] = {(const float*)d_in[2],  (const float*)d_in[10],
                            (const float*)d_in[16], (const float*)d_in[22]};
  const float* conv_b[4] = {(const float*)d_in[3],  (const float*)d_in[11],
                            (const float*)d_in[17], (const float*)d_in[23]};
  const float* g1_w[4]   = {(const float*)d_in[4],  (const float*)d_in[12],
                            (const float*)d_in[18], (const float*)d_in[24]};
  const float* g1_b[4]   = {(const float*)d_in[5],  (const float*)d_in[13],
                            (const float*)d_in[19], (const float*)d_in[25]};
  const float* g2_w[4]   = {(const float*)d_in[6],  (const float*)d_in[14],
                            (const float*)d_in[20], (const float*)d_in[26]};
  const float* g2_b[4]   = {(const float*)d_in[7],  (const float*)d_in[15],
                            (const float*)d_in[21], (const float*)d_in[27]};
  const float* ds_w_f32 = (const float*)d_in[8];
  const float* ds_b     = (const float*)d_in[9];
  const float* lin_in   = (const float*)d_in[28];
  const float* lin_out  = (const float*)d_in[29];
  const float* ae       = (const float*)d_in[30];
  const float* ab       = (const float*)d_in[31];
  const float* emb      = (const float*)d_in[32];
  const float* wq       = (const float*)d_in[33];
  const float* bq       = (const float*)d_in[34];

  // workspace carve-out (all f16 buffers, 256B aligned)
  char* ws = (char*)d_ws;
  size_t off = 0;
  auto carve = [&](size_t bytes) -> void* {
    void* p = ws + off;
    off = (off + bytes + 255) & ~(size_t)255;
    return p;
  };
  _Float16* att  = (_Float16*)carve((size_t)NB * TT_ * FF * 2);
  _Float16* bufA = (_Float16*)carve((size_t)NB * TT_ * DD * 2);
  _Float16* bufB = (_Float16*)carve((size_t)NB * TT_ * DD * 2);
  _Float16* cw[4];
  cw[0] = (_Float16*)carve((size_t)3 * 128 * 32 * 2);
  for (int i = 1; i < 4; ++i) cw[i] = (_Float16*)carve((size_t)3 * 128 * 128 * 2);
  _Float16 *g1h[4], *g2h[4];
  for (int i = 0; i < 4; ++i) {
    g1h[i] = (_Float16*)carve((size_t)128 * 128 * 2);
    g2h[i] = (_Float16*)carve((size_t)128 * 128 * 2);
  }
  _Float16* dsh = (_Float16*)carve((size_t)128 * 32 * 2);

  // ---- weight conversion ----
  conv_w_to_f16<<<(3 * 128 * 32 + 255) / 256, 256, 0, stream>>>(conv_w[0], cw[0], 128, 16, 32);
  for (int i = 1; i < 4; ++i)
    conv_w_to_f16<<<(3 * 128 * 128 + 255) / 256, 256, 0, stream>>>(conv_w[i], cw[i], 128, 128, 128);
  for (int i = 0; i < 4; ++i) {
    mat_to_f16<<<(128 * 128 + 255) / 256, 256, 0, stream>>>(g1_w[i], g1h[i], 128, 128, 128);
    mat_to_f16<<<(128 * 128 + 255) / 256, 256, 0, stream>>>(g2_w[i], g2h[i], 128, 128, 128);
  }
  mat_to_f16<<<(128 * 32 + 255) / 256, 256, 0, stream>>>(ds_w_f32, dsh, 128, 16, 32);

  // ---- self-attention ----
  attn_kernel<<<NB, 256, 0, stream>>>(x, att);

  // ---- TCN (WMMA) ----
  const int layer_grid = NB * (TT_ / 64);
  tcn_layer_kernel<16, 32, 1, true><<<layer_grid, 256, 0, stream>>>(
      att, bufA, cw[0], conv_b[0], g1h[0], g1_b[0], g2h[0], g2_b[0], dsh, ds_b);
  tcn_layer_kernel<128, 128, 2, false><<<layer_grid, 256, 0, stream>>>(
      bufA, bufB, cw[1], conv_b[1], g1h[1], g1_b[1], g2h[1], g2_b[1], nullptr, nullptr);
  tcn_layer_kernel<128, 128, 4, false><<<layer_grid, 256, 0, stream>>>(
      bufB, bufA, cw[2], conv_b[2], g1h[2], g1_b[2], g2h[2], g2_b[2], nullptr, nullptr);
  tcn_layer_kernel<128, 128, 8, false><<<layer_grid, 256, 0, stream>>>(
      bufA, bufB, cw[3], conv_b[3], g1h[3], g1_b[3], g2h[3], g2_b[3], nullptr, nullptr);

  // ---- Hawkes head ----
  hawkes_kernel<<<NB, 256, 0, stream>>>(bufB, idx, emb, wq, bq, lin_in, lin_out,
                                        ae, ab, (float*)d_out);
}
